// FXPBertLayer_50165218017997
// MI455X (gfx1250) — 8-bit matrix operand order repaired and checked by reading, compiled, not run on hardware
//
#include <hip/hip_runtime.h>
#include <hip/hip_bf16.h>


typedef __attribute__((ext_vector_type(8))) int v8i_t;

#define FXP_ONE   65536
#define FXP_HALF  32768

__device__ __forceinline__ int clamp_ll(long long v) {
    const long long hi = 2147483647LL;
    const long long lo = -2147483648LL;
    v = v > hi ? hi : v;
    v = v < lo ? lo : v;
    return (int)v;
}
__device__ __forceinline__ int fxadd(int a, int b) { return clamp_ll((long long)a + (long long)b); }
__device__ __forceinline__ int fxsub(int a, int b) { return clamp_ll((long long)a - (long long)b); }
__device__ __forceinline__ int fxmul(int a, int b) { return clamp_ll(((long long)a * (long long)b) >> 16); }
__device__ __forceinline__ int fxdiv_floor(int a, int b) {
    long long n = ((long long)a) << 16;
    long long d = (long long)b;
    long long q = n / d;
    if ((n % d) != 0 && ((n ^ d) < 0)) q -= 1;
    return clamp_ll(q);
}
__device__ __forceinline__ int imax2(int a, int b) { return a > b ? a : b; }

template <bool SA, bool SB>
__device__ __forceinline__ v8i_t wmma_iu8(v8i_t a, v8i_t b, v8i_t c) {
    return __builtin_amdgcn_wmma_i32_16x16x64_iu8(SA, a, SB, b, c, false, false);
}

__device__ __forceinline__ v8i_t load_afrag(const unsigned char* rowbase, int ksel) {
    v8i_t a;
#pragma unroll
    for (int g = 0; g < 4; ++g) {
        const int2 d = *(const int2*)(rowbase + ((g >> 1) << 5) + (ksel << 4) + ((g & 1) << 3));
        a[2 * g]     = d.x;
        a[2 * g + 1] = d.y;
    }
    return a;
}
__device__ __forceinline__ v8i_t load_bfrag(const unsigned char* rowbase, int ksel) {
    const int4 q0 = *(const int4*)(rowbase + (ksel << 4));
    const int4 q1 = *(const int4*)(rowbase + 32 + (ksel << 4));
    v8i_t b;
    b[0] = q0.x; b[1] = q0.y; b[2] = q0.z; b[3] = q0.w;
    b[4] = q1.x; b[5] = q1.y; b[6] = q1.z; b[7] = q1.w;
    return b;
}

__device__ __forceinline__ long long combine9(const v8i_t* acc, int r) {
    return (long long)acc[0][r]
         + (((long long)acc[1][r] + (long long)acc[3][r]) << 8)
         + (((long long)acc[2][r] + (long long)acc[4][r] + (long long)acc[6][r]) << 16)
         + (((long long)acc[5][r] + (long long)acc[7][r]) << 24)
         + ((long long)acc[8][r] << 32);
}

__global__ __launch_bounds__(256) void fxp_decomp3_kernel(
    const int* __restrict__ src, unsigned char* __restrict__ dst, size_t n)
{
    const size_t nq = n >> 2;
    for (size_t i = blockIdx.x * (size_t)blockDim.x + threadIdx.x; i < nq;
         i += (size_t)gridDim.x * blockDim.x) {
        const int4 xv = ((const int4*)src)[i];
        uchar4 p0, p1, p2;
        p0.x = (unsigned char)(xv.x);       p0.y = (unsigned char)(xv.y);
        p0.z = (unsigned char)(xv.z);       p0.w = (unsigned char)(xv.w);
        p1.x = (unsigned char)(xv.x >> 8);  p1.y = (unsigned char)(xv.y >> 8);
        p1.z = (unsigned char)(xv.z >> 8);  p1.w = (unsigned char)(xv.w >> 8);
        p2.x = (unsigned char)(xv.x >> 16); p2.y = (unsigned char)(xv.y >> 16);
        p2.z = (unsigned char)(xv.z >> 16); p2.w = (unsigned char)(xv.w >> 16);
        ((uchar4*)(dst + 0 * n))[i] = p0;
        ((uchar4*)(dst + 1 * n))[i] = p1;
        ((uchar4*)(dst + 2 * n))[i] = p2;
    }
}

__global__ __launch_bounds__(256) void fxp_decomp2_kernel(
    const int* __restrict__ src, unsigned char* __restrict__ dst, size_t n)
{
    const size_t nq = n >> 2;
    for (size_t i = blockIdx.x * (size_t)blockDim.x + threadIdx.x; i < nq;
         i += (size_t)gridDim.x * blockDim.x) {
        const int4 xv = ((const int4*)src)[i];
        uchar4 p0, p1;
        p0.x = (unsigned char)(xv.x);      p0.y = (unsigned char)(xv.y);
        p0.z = (unsigned char)(xv.z);      p0.w = (unsigned char)(xv.w);
        p1.x = (unsigned char)(xv.x >> 8); p1.y = (unsigned char)(xv.y >> 8);
        p1.z = (unsigned char)(xv.z >> 8); p1.w = (unsigned char)(xv.w >> 8);
        ((uchar4*)(dst + 0 * n))[i] = p0;
        ((uchar4*)(dst + 1 * n))[i] = p1;
    }
}

__global__ __launch_bounds__(256) void fxp_vt_decomp_kernel(
    const int* __restrict__ v, unsigned char* __restrict__ vt,
    int B, int S, int H, int NH)
{
    const size_t TV = (size_t)B * NH * 64 * S;
    for (size_t i = blockIdx.x * (size_t)blockDim.x + threadIdx.x; i < TV;
         i += (size_t)gridDim.x * blockDim.x) {
        const int t    = (int)(i % (size_t)S);
        const size_t r = i / (size_t)S;
        const int dcol = (int)(r % 64);
        const int bh   = (int)(r / 64);
        const int hh = bh % NH, b = bh / NH;
        const int val = v[((size_t)(b * S + t)) * H + hh * 64 + dcol];
        vt[0 * TV + i] = (unsigned char)(val);
        vt[1 * TV + i] = (unsigned char)(val >> 8);
        vt[2 * TV + i] = (unsigned char)(val >> 16);
    }
}

__global__ __launch_bounds__(512) void fxp_gemm_w_kernel(
    const unsigned char* __restrict__ Apl, const unsigned char* __restrict__ Wpl,
    const int* __restrict__ bias, int* __restrict__ out,
    int M, int N, int K)
{
    __shared__ __align__(16) unsigned char Asp[2][3][64][80];
    __shared__ __align__(16) unsigned char Wsp[2][2][64][80];

    const int tid  = threadIdx.x;
    const int lane = tid & 31;
    const int wave = tid >> 5;
    const int wm = wave & 3, wn = wave >> 2;
    const int mblock = blockIdx.y * 64, nblock = blockIdx.x * 64;
    const int lrow = lane & 15, ksel = lane >> 4;
    const size_t MK = (size_t)M * K, NK = (size_t)N * K;
    const int srow = tid >> 3, sc8 = (tid & 7) << 3;

    v8i_t zero = {};
    v8i_t acc[6];
#pragma unroll
    for (int p = 0; p < 6; ++p) acc[p] = zero;

#pragma unroll
    for (int p = 0; p < 3; ++p)
        *(int2*)&Asp[0][p][srow][sc8] =
            *(const int2*)(Apl + p * MK + (size_t)(mblock + srow) * K + sc8);
#pragma unroll
    for (int p = 0; p < 2; ++p)
        *(int2*)&Wsp[0][p][srow][sc8] =
            *(const int2*)(Wpl + p * NK + (size_t)(nblock + srow) * K + sc8);
    __syncthreads();

    const int nk = K >> 6;
    for (int kb = 0; kb < nk; ++kb) {
        const int cur = kb & 1;
        const bool more = (kb + 1) < nk;
        int2 ra[3], rw[2];
        if (more) {
            const int kof = (kb + 1) << 6;
#pragma unroll
            for (int p = 0; p < 3; ++p)
                ra[p] = *(const int2*)(Apl + p * MK + (size_t)(mblock + srow) * K + kof + sc8);
#pragma unroll
            for (int p = 0; p < 2; ++p)
                rw[p] = *(const int2*)(Wpl + p * NK + (size_t)(nblock + srow) * K + kof + sc8);
        }

        v8i_t a[3];
#pragma unroll
        for (int p = 0; p < 3; ++p)
            a[p] = load_afrag(&Asp[cur][p][wm * 16 + lrow][0], ksel);
        {
            const int nrow = wn * 16 + lrow;
            const v8i_t b0 = load_bfrag(&Wsp[cur][0][nrow][0], ksel);
            const v8i_t b1 = load_bfrag(&Wsp[cur][1][nrow][0], ksel);
            acc[0] = wmma_iu8<false, false>(a[0], b0, acc[0]);
            acc[1] = wmma_iu8<false, true >(a[0], b1, acc[1]);
            acc[2] = wmma_iu8<false, false>(a[1], b0, acc[2]);
            acc[3] = wmma_iu8<false, true >(a[1], b1, acc[3]);
            acc[4] = wmma_iu8<true,  false>(a[2], b0, acc[4]);
            acc[5] = wmma_iu8<true,  true >(a[2], b1, acc[5]);
        }

        if (more) {
            __syncthreads();
            const int nxt = cur ^ 1;
#pragma unroll
            for (int p = 0; p < 3; ++p) *(int2*)&Asp[nxt][p][srow][sc8] = ra[p];
#pragma unroll
            for (int p = 0; p < 2; ++p) *(int2*)&Wsp[nxt][p][srow][sc8] = rw[p];
            __syncthreads();
        }
    }

    const int n = nblock + wn * 16 + lrow;
    const int bcol = bias[n];
#pragma unroll
    for (int r = 0; r < 8; ++r) {
        long long s = (long long)acc[0][r]
                    + (((long long)acc[1][r] + (long long)acc[2][r]) << 8)
                    + (((long long)acc[3][r] + (long long)acc[4][r]) << 16)
                    + ((long long)acc[5][r] << 24);
        int val = clamp_ll(s >> 16);
        val = fxadd(val, bcol);
        const int m = mblock + wm * 16 + ksel * 8 + r;
        out[(size_t)m * N + n] = val;
    }
}

__global__ __launch_bounds__(256) void fxp_qk_kernel(
    const unsigned char* __restrict__ qpl, const unsigned char* __restrict__ kpl,
    const int* __restrict__ mask, int* __restrict__ scores,
    int B, int S, int H, int NH)
{
    __shared__ __align__(16) unsigned char Qsp[3][64][80];
    __shared__ __align__(16) unsigned char Ksp[3][32][80];

    const int tid  = threadIdx.x;
    const int lane = tid & 31;
    const int wave = tid >> 5;
    const int wm = wave & 3, wn = wave >> 2;
    const int mblock = blockIdx.y * 64, nblock = blockIdx.x * 32;
    const int bz = blockIdx.z;
    const int hh = bz % NH, b = bz / NH;
    const int lrow = lane & 15, ksel = lane >> 4;
    const size_t MH = (size_t)B * S * H;

    {
        const int srow = tid >> 2, sc16 = (tid & 3) << 4;
#pragma unroll
        for (int p = 0; p < 3; ++p)
            *(int4*)&Qsp[p][srow][sc16] =
                *(const int4*)(qpl + p * MH + (size_t)(b * S + mblock + srow) * H + hh * 64 + sc16);
        if (tid < 128) {
            const int r2 = tid >> 2, c2 = (tid & 3) << 4;
#pragma unroll
            for (int p = 0; p < 3; ++p)
                *(int4*)&Ksp[p][r2][c2] =
                    *(const int4*)(kpl + p * MH + (size_t)(b * S + nblock + r2) * H + hh * 64 + c2);
        }
    }
    __syncthreads();

    v8i_t a[3], bm[3];
#pragma unroll
    for (int p = 0; p < 3; ++p) a[p]  = load_afrag(&Qsp[p][wm * 16 + lrow][0], ksel);
#pragma unroll
    for (int p = 0; p < 3; ++p) bm[p] = load_bfrag(&Ksp[p][wn * 16 + lrow][0], ksel);

    v8i_t zero = {};
    v8i_t acc[9];
#pragma unroll
    for (int p = 0; p < 9; ++p) acc[p] = zero;
    acc[0] = wmma_iu8<false, false>(a[0], bm[0], acc[0]);
    acc[1] = wmma_iu8<false, false>(a[0], bm[1], acc[1]);
    acc[2] = wmma_iu8<false, true >(a[0], bm[2], acc[2]);
    acc[3] = wmma_iu8<false, false>(a[1], bm[0], acc[3]);
    acc[4] = wmma_iu8<false, false>(a[1], bm[1], acc[4]);
    acc[5] = wmma_iu8<false, true >(a[1], bm[2], acc[5]);
    acc[6] = wmma_iu8<true,  false>(a[2], bm[0], acc[6]);
    acc[7] = wmma_iu8<true,  false>(a[2], bm[1], acc[7]);
    acc[8] = wmma_iu8<true,  true >(a[2], bm[2], acc[8]);

    const int n = nblock + wn * 16 + lrow;
    const int mk = mask[b * S + n];
#pragma unroll
    for (int r = 0; r < 8; ++r) {
        const int raw = clamp_ll(combine9(acc, r) >> 16);
        const int scv = fxmul(raw, 8192) + mk;
        const int m = mblock + wm * 16 + ksel * 8 + r;
        scores[((size_t)bz * S + m) * S + n] = scv;
    }
}

__global__ __launch_bounds__(512) void fxp_softmax_kernel(
    const int* __restrict__ scores, unsigned char* __restrict__ apl,
    int S, size_t TA)
{
    __shared__ int sc[512];
    __shared__ long long redl[256];
    __shared__ int redi[256];
    const int t = threadIdx.x;
    const size_t row = blockIdx.x;

    const int s0 = scores[row * S + t];
    sc[t] = s0;
    __syncthreads();
    if (t < 256) redi[t] = imax2(sc[t], sc[t + 256]);
    __syncthreads();
    for (int off = 128; off > 0; off >>= 1) {
        if (t < off) redi[t] = imax2(redi[t], redi[t + off]);
        __syncthreads();
    }
    const int mx = redi[0];

    int z = fxmul(fxsub(s0, mx), 94548);
    if (z < -16 * FXP_ONE) z = -16 * FXP_ONE;
    if (z > 0) z = 0;
    int idx = (int)(((long long)(z + 16 * FXP_ONE) * 255LL) / 1048576LL);
    if (idx < 0) idx = 0;
    if (idx > 255) idx = 255;
    const double g = exp2(-16.0 + (16.0 * (double)idx) / 255.0);
    const int e = clamp_ll(llrint(g * 65536.0));

    sc[t] = e;
    __syncthreads();
    if (t < 256) redl[t] = (long long)sc[t] + (long long)sc[t + 256];
    __syncthreads();
    for (int off = 128; off > 0; off >>= 1) {
        if (t < off) redl[t] += redl[t + off];
        __syncthreads();
    }
    const int attn = clamp_ll((((long long)e) << 16) / redl[0]);

    apl[0 * TA + row * S + t] = (unsigned char)(attn);
    apl[1 * TA + row * S + t] = (unsigned char)(attn >> 8);
    apl[2 * TA + row * S + t] = (unsigned char)(attn >> 16);
}

__global__ __launch_bounds__(256) void fxp_av_kernel(
    const unsigned char* __restrict__ apl, const unsigned char* __restrict__ vtpl,
    int* __restrict__ ctx, int B, int S, int H, int NH)
{
    __shared__ __align__(16) unsigned char Asp[3][64][80];
    __shared__ __align__(16) unsigned char Vsp[3][32][80];

    const int tid  = threadIdx.x;
    const int lane = tid & 31;
    const int wave = tid >> 5;
    const int wm = wave & 3, wn = wave >> 2;
    const int mblock = blockIdx.y * 64, nblock = blockIdx.x * 32;
    const int bz = blockIdx.z;
    const int hh = bz % NH, b = bz / NH;
    const int lrow = lane & 15, ksel = lane >> 4;
    const size_t TA = (size_t)B * NH * S * S;
    const size_t TV = (size_t)B * NH * 64 * S;
    const size_t arow0 = (size_t)bz * S + mblock;
    const size_t vrow0 = (size_t)bz * 64 + nblock;
    const int srow = tid >> 2, sc16 = (tid & 3) << 4;

    v8i_t zero = {};
    v8i_t acc[9];
#pragma unroll
    for (int p = 0; p < 9; ++p) acc[p] = zero;

    const int nk = S >> 6;
    for (int kb = 0; kb < nk; ++kb) {
        __syncthreads();
        const int kof = kb << 6;
#pragma unroll
        for (int p = 0; p < 3; ++p)
            *(int4*)&Asp[p][srow][sc16] =
                *(const int4*)(apl + p * TA + (arow0 + srow) * S + kof + sc16);
        if (tid < 128) {
#pragma unroll
            for (int p = 0; p < 3; ++p)
                *(int4*)&Vsp[p][srow][sc16] =
                    *(const int4*)(vtpl + p * TV + (vrow0 + srow) * S + kof + sc16);
        }
        __syncthreads();

        v8i_t a[3], bm[3];
#pragma unroll
        for (int p = 0; p < 3; ++p) a[p]  = load_afrag(&Asp[p][wm * 16 + lrow][0], ksel);
#pragma unroll
        for (int p = 0; p < 3; ++p) bm[p] = load_bfrag(&Vsp[p][wn * 16 + lrow][0], ksel);

        acc[0] = wmma_iu8<false, false>(a[0], bm[0], acc[0]);
        acc[1] = wmma_iu8<false, false>(a[0], bm[1], acc[1]);
        acc[2] = wmma_iu8<false, true >(a[0], bm[2], acc[2]);
        acc[3] = wmma_iu8<false, false>(a[1], bm[0], acc[3]);
        acc[4] = wmma_iu8<false, false>(a[1], bm[1], acc[4]);
        acc[5] = wmma_iu8<false, true >(a[1], bm[2], acc[5]);
        acc[6] = wmma_iu8<false, false>(a[2], bm[0], acc[6]);
        acc[7] = wmma_iu8<false, false>(a[2], bm[1], acc[7]);
        acc[8] = wmma_iu8<false, true >(a[2], bm[2], acc[8]);
    }

    const int n = nblock + wn * 16 + lrow;
#pragma unroll
    for (int r = 0; r < 8; ++r) {
        const int m = mblock + wm * 16 + ksel * 8 + r;
        ctx[((size_t)(b * S + m)) * H + hh * 64 + n] = clamp_ll(combine9(acc, r) >> 16);
    }
}

__global__ __launch_bounds__(256) void fxp_add_ln_kernel(
    const int* __restrict__ xa, const int* __restrict__ xb,
    const int* __restrict__ gamma, const int* __restrict__ beta,
    int* __restrict__ out, int H)
{
    __shared__ long long red[256];
    const int row = blockIdx.x;
    const int t   = threadIdx.x;

    int xr[3];
    long long s = 0;
#pragma unroll
    for (int i = 0; i < 3; ++i) {
        const int j = t + 256 * i;
        const int val = fxadd(xa[(size_t)row * H + j], xb[(size_t)row * H + j]);
        xr[i] = val;
        s += (long long)val;
    }
    red[t] = s;
    __syncthreads();
    for (int off = 128; off > 0; off >>= 1) {
        if (t < off) red[t] += red[t + off];
        __syncthreads();
    }
    const int mean = clamp_ll((red[0] * 85LL) >> 16);
    __syncthreads();

    int xc[3];
    long long vs = 0;
#pragma unroll
    for (int i = 0; i < 3; ++i) {
        xc[i] = fxsub(xr[i], mean);
        vs += (((long long)xc[i] * (long long)xc[i]) >> 16);
    }
    red[t] = vs;
    __syncthreads();
    for (int off = 128; off > 0; off >>= 1) {
        if (t < off) red[t] += red[t + off];
        __syncthreads();
    }
    const int var_eps = clamp_ll((red[0] * 85LL) >> 16);

    const double xf = (double)var_eps / 65536.0;
    int y = clamp_ll(llrint((1.0 / sqrt(xf + 1e-12)) * 65536.0));
#pragma unroll
    for (int it = 0; it < 2; ++it) {
        const int xy2 = fxmul(var_eps, fxmul(y, y));
        y = fxmul(y, fxsub(98304  , fxmul(FXP_HALF, xy2)));
    }

#pragma unroll
    for (int i = 0; i < 3; ++i) {
        const int j = t + 256 * i;
        out[(size_t)row * H + j] = fxadd(fxmul(fxmul(xc[i], y), gamma[j]), beta[j]);
    }
}

__global__ __launch_bounds__(256) void fxp_gelu_kernel(int* __restrict__ data, long long n)
{
    const int C0  = 52290;
    const int C1  = 2930;
    const int N27 = 27 * FXP_ONE;
    const int N9  = 9 * FXP_ONE;
    for (long long i = blockIdx.x * (long long)blockDim.x + threadIdx.x;
         i < n; i += (long long)gridDim.x * blockDim.x) {
        const int x = data[i];
        const int x3 = fxmul(fxmul(x, x), x);
        const int inner = fxmul(C0, fxadd(x, fxmul(C1, x3)));
        const int inner2 = fxmul(inner, inner);
        const int tq = fxdiv_floor(fxmul(inner, fxadd(N27, inner2)),
                                   fxadd(N27, fxmul(N9, inner2)));
        data[i] = fxmul(fxmul(FXP_HALF, x), fxadd(FXP_ONE, tq));
    }
}

extern "C" void kernel_launch(void* const* d_in, const int* in_sizes, int n_in,
                              void* d_out, int out_size, void* d_ws, size_t ws_size,
                              hipStream_t stream) {
    (void)in_sizes; (void)n_in; (void)out_size; (void)ws_size;
    const int B = 8, S = 512, H = 768, DFF = 3072, NH = 12;
    const int M = B * S;

    const int* x    = (const int*)d_in[0];
    const int* mask = (const int*)d_in[1];
    const int* wq = (const int*)d_in[2];  const int* bq = (const int*)d_in[3];
    const int* wk = (const int*)d_in[4];  const int* bk = (const int*)d_in[5];
    const int* wv = (const int*)d_in[6];  const int* bv = (const int*)d_in[7];
    const int* wo = (const int*)d_in[8];  const int* bo = (const int*)d_in[9];
    const int* ln1g = (const int*)d_in[10]; const int* ln1b = (const int*)d_in[11];
    const int* w1 = (const int*)d_in[12]; const int* b1 = (const int*)d_in[13];
    const int* w2 = (const int*)d_in[14]; const int* b2 = (const int*)d_in[15];
    const int* ln2g = (const int*)d_in[16]; const int* ln2b = (const int*)d_in[17];

    const size_t elH  = (size_t)M * H;
    const size_t elFF = (size_t)M * DFF;
    const size_t TA   = (size_t)B * NH * S * S;
    const size_t TV   = (size_t)B * NH * 64 * S;
    const size_t maxWK = (size_t)DFF * H;

    char* ws = (char*)d_ws;
    size_t off = 0;
    int* qb   = (int*)(ws + off); off += elH * 4;
    int* kb   = (int*)(ws + off); off += elH * 4;
    int* vb   = (int*)(ws + off); off += elH * 4;
    int* ctxb = (int*)(ws + off); off += elH * 4;
    int* tmpb = (int*)(ws + off); off += elH * 4;
    int* x1b  = (int*)(ws + off); off += elH * 4;
    unsigned char* xpl  = (unsigned char*)(ws + off); off += 3 * elH;
    unsigned char* wpl  = (unsigned char*)(ws + off); off += 2 * maxWK;
    unsigned char* qpl  = (unsigned char*)(ws + off); off += 3 * elH;
    unsigned char* kpl  = (unsigned char*)(ws + off); off += 3 * elH;
    unsigned char* vtpl = (unsigned char*)(ws + off); off += 3 * TV;
    char* regE = ws + off; off += (TA * 4 > 3 * elFF ? TA * 4 : 3 * elFF);
    int* scores = (int*)regE;
    unsigned char* aplBig = (unsigned char*)regE;
    char* regF = ws + off; off += (3 * TA > elFF * 4 ? 3 * TA : elFF * 4);
    unsigned char* attnpl = (unsigned char*)regF;
    int* ff1b = (int*)regF;

    const dim3 blk(256);
    const dim3 blk512(512);
    const dim3 gH(H / 64, M / 64);
    const dim3 gF(DFF / 64, M / 64);
    const dim3 gQK(S / 32, S / 64, B * NH);
    const dim3 gAV(64 / 32, S / 64, B * NH);

    fxp_decomp3_kernel<<<2048, blk, 0, stream>>>(x, xpl, elH);
    fxp_decomp2_kernel<<<1024, blk, 0, stream>>>(wq, wpl, (size_t)H * H);
    fxp_gemm_w_kernel<<<gH, blk512, 0, stream>>>(xpl, wpl, bq, qb, M, H, H);
    fxp_decomp2_kernel<<<1024, blk, 0, stream>>>(wk, wpl, (size_t)H * H);
    fxp_gemm_w_kernel<<<gH, blk512, 0, stream>>>(xpl, wpl, bk, kb, M, H, H);
    fxp_decomp2_kernel<<<1024, blk, 0, stream>>>(wv, wpl, (size_t)H * H);
    fxp_gemm_w_kernel<<<gH, blk512, 0, stream>>>(xpl, wpl, bv, vb, M, H, H);

    fxp_decomp3_kernel<<<2048, blk, 0, stream>>>(qb, qpl, elH);
    fxp_decomp3_kernel<<<2048, blk, 0, stream>>>(kb, kpl, elH);
    fxp_vt_decomp_kernel<<<4096, blk, 0, stream>>>(vb, vtpl, B, S, H, NH);
    fxp_qk_kernel<<<gQK, blk, 0, stream>>>(qpl, kpl, mask, scores, B, S, H, NH);
    fxp_softmax_kernel<<<B * NH * S, blk512, 0, stream>>>(scores, attnpl, S, TA);
    fxp_av_kernel<<<gAV, blk, 0, stream>>>(attnpl, vtpl, ctxb, B, S, H, NH);

    fxp_decomp3_kernel<<<2048, blk, 0, stream>>>(ctxb, xpl, elH);
    fxp_decomp2_kernel<<<1024, blk, 0, stream>>>(wo, wpl, (size_t)H * H);
    fxp_gemm_w_kernel<<<gH, blk512, 0, stream>>>(xpl, wpl, bo, tmpb, M, H, H);
    fxp_add_ln_kernel<<<M, blk, 0, stream>>>(x, tmpb, ln1g, ln1b, x1b, H);

    fxp_decomp3_kernel<<<2048, blk, 0, stream>>>(x1b, xpl, elH);
    fxp_decomp2_kernel<<<2048, blk, 0, stream>>>(w1, wpl, (size_t)DFF * H);
    fxp_gemm_w_kernel<<<gF, blk512, 0, stream>>>(xpl, wpl, b1, ff1b, M, DFF, H);
    fxp_gelu_kernel<<<4096, blk, 0, stream>>>(ff1b, (long long)elFF);
    fxp_decomp3_kernel<<<4096, blk, 0, stream>>>(ff1b, aplBig, elFF);
    fxp_decomp2_kernel<<<2048, blk, 0, stream>>>(w2, wpl, (size_t)H * DFF);
    fxp_gemm_w_kernel<<<gH, blk512, 0, stream>>>(aplBig, wpl, b2, tmpb, M, H, DFF);

    fxp_add_ln_kernel<<<M, blk, 0, stream>>>(x1b, tmpb, ln2g, ln2b, (int*)d_out, H);
}
